// GraphConv_32031866093870
// MI455X (gfx1250) — matrix operand order repaired and checked by reading, not compiled, not run on hardware
//
#include <hip/hip_runtime.h>

// ---------------------------------------------------------------------------
// CDNA5 (gfx1250) GNN pipeline: WMMA bf16 GEMMs + scatter-max fusion.
// Round 3: weights stored transposed (Wt[N][K] bf16) so every B-fragment is a
// contiguous 64B lane slice -> global_load_b128 feeding back-to-back v_wmma;
// edge GEMM A-side branch lifted to (wave-uniform) step level for the same.
// ---------------------------------------------------------------------------

typedef __attribute__((ext_vector_type(16))) __bf16 v16bf;
typedef __attribute__((ext_vector_type(8)))  float  v8f;

#define NNODES 16384
#define NEDGES 524288
#define KNN_K  16
#define CHUNK_T 2048                 // dynamic-layer targets per chunk
#define CHUNK_ROWS (CHUNK_T * 17)    // 34816 edge rows per chunk
#define ENC_NEGINF 0x007FFFFFu

__device__ __forceinline__ float lrelu(float v, float s) { return v >= 0.f ? v : s * v; }

// monotonic float -> uint encoding so atomicMax(uint) == float max
__device__ __forceinline__ unsigned encf(float f) {
  unsigned u = __float_as_uint(f);
  return (u & 0x80000000u) ? ~u : (u | 0x80000000u);
}
__device__ __forceinline__ float decf(unsigned u) {
  unsigned b = (u & 0x80000000u) ? (u ^ 0x80000000u) : ~u;
  return __uint_as_float(b);
}

// --- WMMA fragment loaders (ISA 7.12.2 layouts, wave32) --------------------
// A: 16x32 bf16, from f32 row. lane half h: V0..3 -> K=2v+8h, V4..7 -> K=16+2(v-4)+8h
__device__ __forceinline__ v16bf load_a_frag_row(const float* Ar, int k0, int half) {
  v16bf a;
#pragma unroll
  for (int v = 0; v < 8; ++v) {
    int kb = k0 + ((v < 4) ? 2 * v : 16 + 2 * (v - 4)) + 8 * half;
    a[2 * v]     = (__bf16)Ar[kb];
    a[2 * v + 1] = (__bf16)Ar[kb + 1];
  }
  return a;
}
// A: same layout, from a bf16 row (no conversion, contiguous 64B per lane)
__device__ __forceinline__ v16bf load_a_frag_row_h(const __bf16* Ar, int k0, int half) {
  v16bf a;
#pragma unroll
  for (int v = 0; v < 8; ++v) {
    int kb = k0 + ((v < 4) ? 2 * v : 16 + 2 * (v - 4)) + 8 * half;
    a[2 * v]     = Ar[kb];
    a[2 * v + 1] = Ar[kb + 1];
  }
  return a;
}
// B: 32x16. Wt is the TRANSPOSED weight, N x K row-major bf16, so a lane's
// fragment (col fixed, K-slice) is contiguous:  Wt[col][k0 .. k0+31].
__device__ __forceinline__ v16bf load_b_frag(const __bf16* Wt, int K, int k0, int c0,
                                             int lane) {
  v16bf b;
  const __bf16* Br = Wt + (size_t)(c0 + (lane & 15)) * K;
  int half = lane >> 4;
#pragma unroll
  for (int v = 0; v < 8; ++v) {
    int kk = k0 + ((v < 4) ? 2 * v : 16 + 2 * (v - 4)) + 8 * half;
    b[2 * v]     = Br[kk];
    b[2 * v + 1] = Br[kk + 1];
  }
  return b;
}

// one 16x16x32 WMMA step, A from f32
__device__ __forceinline__ void gemm_step(v8f& acc, const float* Ar, const __bf16* Wt,
                                          int K, int k0, int c0, int lane, int half) {
  v16bf a = load_a_frag_row(Ar, k0, half);
  v16bf b = load_b_frag(Wt, K, k0, c0, lane);
  acc = __builtin_amdgcn_wmma_f32_16x16x32_bf16(false, a, false, b, (short)0, acc, false,
                                                false);
}
// one step, A from bf16
__device__ __forceinline__ void gemm_step_h(v8f& acc, const __bf16* Ar, const __bf16* Wt,
                                            int K, int k0, int c0, int lane, int half) {
  v16bf a = load_a_frag_row_h(Ar, k0, half);
  v16bf b = load_b_frag(Wt, K, k0, c0, lane);
  acc = __builtin_amdgcn_wmma_f32_16x16x32_bf16(false, a, false, b, (short)0, acc, false,
                                                false);
}

// ---------------------------------------------------------------------------
// elementwise helpers
// ---------------------------------------------------------------------------
// fp32 W[K][N] -> bf16 Wt[N][K] (one-time, read-coalesced)
__global__ void k_cvt_bf16_t(const float* __restrict__ s, __bf16* __restrict__ d, int K,
                             int N) {
  int g = blockIdx.x * 256 + threadIdx.x;
  if (g >= K * N) return;
  int k = g / N, n = g - k * N;
  d[(size_t)n * K + k] = (__bf16)s[g];
}

__global__ void k_fill_u32(unsigned* p, long long n, unsigned v) {
  long long g = (long long)blockIdx.x * blockDim.x + threadIdx.x;
  if (g < n) p[g] = v;
}

__global__ void k_decode_max(unsigned* p, long long n) {
  long long g = (long long)blockIdx.x * blockDim.x + threadIdx.x;
  if (g >= n) return;
  unsigned u = p[g];
  float f = (u == ENC_NEGINF) ? 0.f : decf(u);   // empty segment -> 0
  ((float*)p)[g] = f;
}

__global__ void k_copy_strided(const float* src, int lds, float* dst, int ldd, int C,
                               long long total) {
  long long g = (long long)blockIdx.x * blockDim.x + threadIdx.x;
  if (g >= total) return;
  int c = (int)(g % C);
  long long r = g / C;
  dst[r * (long long)ldd + c] = src[r * (long long)lds + c];
}

// ---------------------------------------------------------------------------
// static-layer edge message + scatter-max:  H[dst] = max(cat(x_src - x_dst, ea))
// ---------------------------------------------------------------------------
__global__ void k_edge_scatter(const float* __restrict__ x, int lda, int Dx,
                               const float* __restrict__ ea,
                               const int* __restrict__ srcI, const int* __restrict__ dstI,
                               long long E, unsigned* __restrict__ H, int C) {
  long long g = (long long)blockIdx.x * blockDim.x + threadIdx.x;
  long long total = E * C;
  if (g >= total) return;
  int c = (int)(g % C);
  long long e = g / C;
  int s = srcI[e], d = dstI[e];
  float v;
  if (c < Dx) v = x[(size_t)s * lda + c] - x[(size_t)d * lda + c];
  else        v = ea[(size_t)e * 64 + (c - Dx)];
  atomicMax(&H[(size_t)d * C + c], encf(v));
}

// ---------------------------------------------------------------------------
// row L2-normalize (cosine kNN prep): one wave32 per row; emits bf16 directly
// ---------------------------------------------------------------------------
__global__ void k_rownorm(const float* __restrict__ x, int lda, int D,
                          __bf16* __restrict__ xnb, int n) {
  int lane = threadIdx.x & 31;
  int row = blockIdx.x * (blockDim.x >> 5) + (threadIdx.x >> 5);
  if (row >= n) return;
  float s = 0.f;
  for (int c = lane; c < D; c += 32) {
    float v = x[(size_t)row * lda + c];
    s += v * v;
  }
#pragma unroll
  for (int off = 16; off > 0; off >>= 1) s += __shfl_xor(s, off, 32);
  float r = rsqrtf(s + 1e-12f);
  for (int c = lane; c < D; c += 32)
    xnb[(size_t)row * D + c] = (__bf16)(x[(size_t)row * lda + c] * r);
}

// ---------------------------------------------------------------------------
// kNN: 16 rows per block; 8 waves compute 16x16 WMMA sim tiles (128 cols/sweep),
// streamed through LDS into per-row sorted top-16 lists. Diagonal masked -inf.
// ---------------------------------------------------------------------------
__global__ void k_knn(const __bf16* __restrict__ xn, int D, int n, int* __restrict__ knn) {
  __shared__ float simLDS[16][128];
  __shared__ float tv[16][16];
  __shared__ int   ti[16][16];
  int tid = threadIdx.x, lane = tid & 31, w = tid >> 5;
  int i0 = blockIdx.x * 16;
  if (tid < 16) {
#pragma unroll
    for (int q = 0; q < 16; ++q) { tv[tid][q] = -__builtin_inff(); ti[tid][q] = 0; }
  }
  __syncthreads();
  int half = lane >> 4;
  const __bf16* Ar = xn + (size_t)(i0 + (lane & 15)) * D;
  int sweeps = n >> 7;
  for (int s = 0; s < sweeps; ++s) {
    int j0 = s * 128 + w * 16;
    const __bf16* Br = xn + (size_t)(j0 + (lane & 15)) * D;  // B = xn^T
    v8f acc = {};
    for (int k0 = 0; k0 < D; k0 += 64) {       // D % 64 == 0
#pragma unroll
      for (int u = 0; u < 2; ++u) {
        int kq = k0 + 32 * u;
        v16bf a = load_a_frag_row_h(Ar, kq, half);
        v16bf b;
#pragma unroll
        for (int v = 0; v < 8; ++v) {
          int kk = kq + ((v < 4) ? 2 * v : 16 + 2 * (v - 4)) + 8 * half;
          b[2 * v]     = Br[kk];
          b[2 * v + 1] = Br[kk + 1];
        }
        acc = __builtin_amdgcn_wmma_f32_16x16x32_bf16(false, a, false, b, (short)0, acc,
                                                      false, false);
      }
    }
    int colL = w * 16 + (lane & 15);
    int gcol = s * 128 + colL;
#pragma unroll
    for (int v = 0; v < 8; ++v) {
      int r = v + 8 * half;
      simLDS[r][colL] = (gcol == i0 + r) ? -__builtin_inff() : acc[v];
    }
    __syncthreads();
    if (tid < 16) {                       // one thread maintains its row's top-16
      float worst = tv[tid][15];
      for (int c = 0; c < 128; ++c) {
        float v = simLDS[tid][c];
        if (v > worst) {
          int p = 15;
          while (p > 0 && tv[tid][p - 1] < v) {
            tv[tid][p] = tv[tid][p - 1];
            ti[tid][p] = ti[tid][p - 1];
            --p;
          }
          tv[tid][p] = v;
          ti[tid][p] = s * 128 + c;
          worst = tv[tid][15];
        }
      }
    }
    __syncthreads();
  }
  if (tid < 16) {
#pragma unroll
    for (int q = 0; q < 16; ++q) knn[(size_t)(i0 + tid) * KNN_K + q] = ti[tid][q];
  }
}

// ---------------------------------------------------------------------------
// generic node GEMM: out = epilogue(A[16-tile] @ W).  EPI 0: lrelu, 1: lrelu+aux,
// 2: plain.  Block = 256 thr (8 waves), tile = 16 rows x 128 cols. Wt bf16 [N][K].
// ---------------------------------------------------------------------------
template <int EPI>
__global__ void k_gemm(const float* __restrict__ A, int lda, int K,
                       const __bf16* __restrict__ Wt, int N,
                       float* __restrict__ out, int ldo, float slope,
                       const float* __restrict__ aux, int ldaux) {
  int lane = threadIdx.x & 31, w = threadIdx.x >> 5;
  int r0 = blockIdx.x * 16;
  int c0 = blockIdx.y * 128 + w * 16;
  if (c0 >= N) return;                               // wave-uniform
  int half = lane >> 4;
  const float* Ar = A + (size_t)(r0 + (lane & 15)) * lda;
  v8f acc = {};
  int k0 = 0;
  for (; k0 + 64 <= K; k0 += 64) {                   // 2 back-to-back WMMAs
    if (k0 + 64 < K)
      __builtin_prefetch(&Wt[(size_t)(c0 + (lane & 15)) * K + k0 + 64], 0, 1);
    gemm_step(acc, Ar, Wt, K, k0, c0, lane, half);
    gemm_step(acc, Ar, Wt, K, k0 + 32, c0, lane, half);
  }
  if (k0 < K) gemm_step(acc, Ar, Wt, K, k0, c0, lane, half);  // K % 64 == 32 tail
  int col = c0 + (lane & 15);
#pragma unroll
  for (int v = 0; v < 8; ++v) {
    size_t r = (size_t)(r0 + v + 8 * half);
    float val = acc[v];
    if (EPI == 0)      out[r * ldo + col] = lrelu(val, slope);
    else if (EPI == 1) out[r * ldo + col] = lrelu(val, slope) + aux[r * ldaux + col];
    else               out[r * ldo + col] = val;
  }
}

// ---------------------------------------------------------------------------
// dynamic-layer edge GEMM #1: A row r (target i=r/17, t=r%17) is
// cat(x_i, x_src - x_i); out = lrelu(A @ m1, 0.2) stored bf16.
// Dx % 32 == 0 -> each 32-wide K-step is entirely x_i or entirely (x_s - x_i):
// branch is wave-uniform and both paths are contiguous (b128-vectorizable).
// ---------------------------------------------------------------------------
__device__ __forceinline__ void edge_step(v8f& acc, const float* xi, const float* xs,
                                          int Dx, const __bf16* Wt, int K, int k0,
                                          int c0, int lane, int half) {
  v16bf a;
  if (k0 < Dx) {
#pragma unroll
    for (int v = 0; v < 8; ++v) {
      int kb = k0 + ((v < 4) ? 2 * v : 16 + 2 * (v - 4)) + 8 * half;
      a[2 * v]     = (__bf16)xi[kb];
      a[2 * v + 1] = (__bf16)xi[kb + 1];
    }
  } else {
#pragma unroll
    for (int v = 0; v < 8; ++v) {
      int kb = k0 - Dx + ((v < 4) ? 2 * v : 16 + 2 * (v - 4)) + 8 * half;
      a[2 * v]     = (__bf16)(xs[kb] - xi[kb]);
      a[2 * v + 1] = (__bf16)(xs[kb + 1] - xi[kb + 1]);
    }
  }
  v16bf b = load_b_frag(Wt, K, k0, c0, lane);
  acc = __builtin_amdgcn_wmma_f32_16x16x32_bf16(false, a, false, b, (short)0, acc, false,
                                                false);
}

__global__ void k_gemm_edge(const float* __restrict__ x, int lda, int Dx,
                            const int* __restrict__ knn, int t0,
                            const __bf16* __restrict__ Wt, int N,
                            __bf16* __restrict__ Me, float slope) {
  int lane = threadIdx.x & 31, w = threadIdx.x >> 5;
  int r0 = blockIdx.x * 16;
  int c0 = blockIdx.y * 128 + w * 16;
  if (c0 >= N) return;
  int half = lane >> 4;
  int er = r0 + (lane & 15);
  int i = t0 + er / 17;
  int t = er % 17;
  int srcn = (t < KNN_K) ? knn[(size_t)i * KNN_K + t] : i;   // self loop last
  const float* xi = x + (size_t)i * lda;
  const float* xs = x + (size_t)srcn * lda;
  int K = 2 * Dx;                                             // 128 or 256
  v8f acc = {};
  for (int k0 = 0; k0 < K; k0 += 64) {                        // K % 64 == 0
    edge_step(acc, xi, xs, Dx, Wt, K, k0, c0, lane, half);
    edge_step(acc, xi, xs, Dx, Wt, K, k0 + 32, c0, lane, half);
  }
  int col = c0 + (lane & 15);
#pragma unroll
  for (int v = 0; v < 8; ++v) {
    size_t r = (size_t)(r0 + v + 8 * half);
    Me[r * N + col] = (__bf16)lrelu(acc[v], slope);
  }
}

// dynamic-layer edge GEMM #2 fused with segment-max: scatter into H[target].
// A (= Me) is bf16, K is 128 or 192.
__global__ void k_gemm_scatter(const __bf16* __restrict__ Me, int K,
                               const __bf16* __restrict__ Wt, int N,
                               unsigned* __restrict__ H, int t0, float slope) {
  int lane = threadIdx.x & 31, w = threadIdx.x >> 5;
  int r0 = blockIdx.x * 16;
  int c0 = blockIdx.y * 128 + w * 16;
  if (c0 >= N) return;
  int half = lane >> 4;
  const __bf16* Ar = Me + (size_t)(r0 + (lane & 15)) * K;
  v8f acc = {};
  int k0 = 0;
  for (; k0 + 64 <= K; k0 += 64) {
    gemm_step_h(acc, Ar, Wt, K, k0, c0, lane, half);
    gemm_step_h(acc, Ar, Wt, K, k0 + 32, c0, lane, half);
  }
  if (k0 < K) gemm_step_h(acc, Ar, Wt, K, k0, c0, lane, half);  // K % 64 == 32 tail
  int col = c0 + (lane & 15);
#pragma unroll
  for (int v = 0; v < 8; ++v) {
    int er = r0 + v + 8 * half;
    int i = t0 + er / 17;
    atomicMax(&H[(size_t)i * N + col], encf(lrelu(acc[v], slope)));
  }
}

// ---------------------------------------------------------------------------
// BatchNorm1d (training: biased batch stats, eps=1e-5): per-column stats then
// fused scale/shift + LeakyReLU apply.
// ---------------------------------------------------------------------------
__global__ void k_bn_stats(const float* __restrict__ Y, int ld, int n,
                           const float* __restrict__ g, const float* __restrict__ b,
                           float* __restrict__ scale, float* __restrict__ shift) {
  int c = blockIdx.x;
  __shared__ float s1[256], s2[256];
  float a = 0.f, q = 0.f;
  for (int r = threadIdx.x; r < n; r += 256) {
    float v = Y[(size_t)r * ld + c];
    a += v;
    q += v * v;
  }
  s1[threadIdx.x] = a;
  s2[threadIdx.x] = q;
  __syncthreads();
  for (int off = 128; off > 0; off >>= 1) {
    if (threadIdx.x < off) {
      s1[threadIdx.x] += s1[threadIdx.x + off];
      s2[threadIdx.x] += s2[threadIdx.x + off];
    }
    __syncthreads();
  }
  if (threadIdx.x == 0) {
    float m = s1[0] / n;
    float var = s2[0] / n - m * m;
    float sc = g[c] * rsqrtf(var + 1e-5f);
    scale[c] = sc;
    shift[c] = b[c] - m * sc;
  }
}

__global__ void k_bn_apply(float* __restrict__ Y, int ld, int C, long long total,
                           const float* __restrict__ sc, const float* __restrict__ sh,
                           float slope) {
  long long g = (long long)blockIdx.x * blockDim.x + threadIdx.x;
  if (g >= total) return;
  int c = (int)(g % C);
  long long r = g / C;
  float v = Y[r * (long long)ld + c] * sc[c] + sh[c];
  Y[r * (long long)ld + c] = lrelu(v, slope);
}

// ---------------------------------------------------------------------------
// host orchestration
// ---------------------------------------------------------------------------
extern "C" void kernel_launch(void* const* d_in, const int* in_sizes, int n_in,
                              void* d_out, int out_size, void* d_ws, size_t ws_size,
                              hipStream_t stream) {
  (void)in_sizes; (void)n_in; (void)out_size; (void)ws_size;
  const int N_ = NNODES;
  const long long E_ = NEDGES;

  const float* node = (const float*)d_in[0];
  const float* ea   = (const float*)d_in[1];
  const int*   ei   = (const int*)d_in[2];
  const int* srcI = ei;
  const int* dstI = ei + E_;
  const float* f1g = (const float*)d_in[23];
  const float* f1b = (const float*)d_in[24];
  const float* f2g1 = (const float*)d_in[26];
  const float* f2b1 = (const float*)d_in[27];
  const float* f2g2 = (const float*)d_in[29];
  const float* f2b2 = (const float*)d_in[30];

  // workspace layout (byte-based), ~272 MB peak
  char* base = (char*)d_ws;
  size_t off = 0;
  auto allocB = [&](size_t bytes) {
    void* p = base + off;
    off = (off + bytes + 255) & ~(size_t)255;
    return p;
  };
  float* cat1 = (float*)allocB((size_t)N_ * 704 * 4);   // [sg1|dg1|dg2|sg2|sg3]
  float* cat2 = (float*)allocB((size_t)N_ * 1728 * 4);  // [f1|cat1]
  __bf16* xnb = (__bf16*)allocB((size_t)N_ * 128 * 2);
  int*   knn  = (int*)allocB((size_t)N_ * KNN_K * 4);
  unsigned* H = (unsigned*)allocB((size_t)N_ * 192 * 4);  // scatter-max accumulator
  float* T1 = (float*)allocB((size_t)N_ * 192 * 4);
  float* T2 = (float*)allocB((size_t)N_ * 192 * 4);
  __bf16* Me = (__bf16*)allocB((size_t)CHUNK_ROWS * 192 * 2);
  float* h1 = (float*)allocB((size_t)N_ * 512 * 4);
  float* h2 = (float*)allocB((size_t)N_ * 256 * 4);
  float* bnsc = (float*)allocB(1728 * 4);
  float* bnsh = (float*)allocB(1728 * 4);
  __bf16* wpool = (__bf16*)allocB((size_t)3 * 1024 * 1024 * 2);

  auto ew = [](long long tot) { return dim3((unsigned)((tot + 255) / 256)); };

  // one-time fp32 W[K][N] -> bf16 Wt[N][K] conversion into wpool
  size_t woff = 0;
  auto cvtW = [&](int idx, int K, int N) {
    __bf16* d = wpool + woff;
    size_t n = (size_t)K * N;
    woff += (n + 127) & ~(size_t)127;
    k_cvt_bf16_t<<<ew((long long)n), 256, 0, stream>>>((const float*)d_in[idx], d, K, N);
    return (const __bf16*)d;
  };
  const __bf16* sg1w1 = cvtW(4, 128, 96);
  const __bf16* sg1w2 = cvtW(5, 96, 64);
  const __bf16* sg1w3 = cvtW(6, 64, 64);
  const __bf16* sg1l  = cvtW(7, 64, 64);
  const __bf16* sg2w1 = cvtW(8, 128, 128);
  const __bf16* sg2w2 = cvtW(9, 128, 128);
  const __bf16* sg2w3 = cvtW(10, 128, 128);
  const __bf16* sg2l  = cvtW(11, 64, 128);
  const __bf16* sg3w1 = cvtW(12, 192, 192);
  const __bf16* sg3w2 = cvtW(13, 192, 192);
  const __bf16* sg3w3 = cvtW(14, 192, 192);
  const __bf16* sg3l  = cvtW(15, 128, 192);
  const __bf16* dg1m1 = cvtW(16, 128, 128);
  const __bf16* dg1m2 = cvtW(17, 128, 128);
  const __bf16* dg1l  = cvtW(18, 64, 128);
  const __bf16* dg2m1 = cvtW(19, 256, 192);
  const __bf16* dg2m2 = cvtW(20, 192, 192);
  const __bf16* dg2l  = cvtW(21, 128, 192);
  const __bf16* f1w   = cvtW(22, 704, 1024);
  const __bf16* f2w1  = cvtW(25, 1728, 512);
  const __bf16* f2w2  = cvtW(28, 512, 256);
  const __bf16* f2w3  = cvtW(31, 256, 128);

  // ---- static edge-conv layer ----
  auto static_layer = [&](const float* x, int lda, int Dx, const __bf16* w1, int c1,
                          const __bf16* w2, int c2, const __bf16* w3, int c3,
                          const __bf16* wl, float* outp) {
    int C = Dx + 64;
    long long nh = (long long)N_ * C;
    k_fill_u32<<<ew(nh), 256, 0, stream>>>(H, nh, ENC_NEGINF);
    k_edge_scatter<<<ew(E_ * C), 256, 0, stream>>>(x, lda, Dx, ea, srcI, dstI, E_, H, C);
    k_decode_max<<<ew(nh), 256, 0, stream>>>(H, nh);
    k_gemm<0><<<dim3(N_ / 16, (c1 + 127) / 128), 256, 0, stream>>>(
        (const float*)H, C, C, w1, c1, T1, c1, 0.1f, nullptr, 0);
    k_gemm<0><<<dim3(N_ / 16, (c2 + 127) / 128), 256, 0, stream>>>(
        T1, c1, c1, w2, c2, T2, c2, 0.1f, nullptr, 0);
    k_gemm<0><<<dim3(N_ / 16, (c3 + 127) / 128), 256, 0, stream>>>(
        T2, c2, c2, w3, c3, T1, c3, 0.1f, nullptr, 0);
    k_gemm<1><<<dim3(N_ / 16, (c3 + 127) / 128), 256, 0, stream>>>(
        x, lda, Dx, wl, c3, outp, 704, 0.1f, T1, c3);
  };

  // ---- dynamic (kNN) edge-conv layer ----
  auto dynamic_layer = [&](const float* x, int lda, int Dx, const __bf16* m1, int c1,
                           const __bf16* m2, int c2, const __bf16* wl, float* outp) {
    k_rownorm<<<dim3(N_ / 8), 256, 0, stream>>>(x, lda, Dx, xnb, N_);
    k_knn<<<dim3(N_ / 16), 256, 0, stream>>>(xnb, Dx, N_, knn);
    long long nh = (long long)N_ * c2;
    k_fill_u32<<<ew(nh), 256, 0, stream>>>(H, nh, ENC_NEGINF);
    for (int ch = 0; ch < N_ / CHUNK_T; ++ch) {
      int t0 = ch * CHUNK_T;
      k_gemm_edge<<<dim3(CHUNK_ROWS / 16, (c1 + 127) / 128), 256, 0, stream>>>(
          x, lda, Dx, knn, t0, m1, c1, Me, 0.2f);
      k_gemm_scatter<<<dim3(CHUNK_ROWS / 16, (c2 + 127) / 128), 256, 0, stream>>>(
          Me, c1, m2, c2, H, t0, 0.2f);
    }
    k_decode_max<<<ew(nh), 256, 0, stream>>>(H, nh);
    k_gemm<1><<<dim3(N_ / 16, (c2 + 127) / 128), 256, 0, stream>>>(
        x, lda, Dx, wl, c2, outp, 704, 0.1f, (const float*)H, c2);
  };

  // sg1 -> sg2 -> sg3, dg1(sg1), dg2(dg1); outputs written strided into cat1
  static_layer(node, 64, 64, sg1w1, 96, sg1w2, 64, sg1w3, 64, sg1l, cat1 + 0);
  static_layer(cat1 + 0, 704, 64, sg2w1, 128, sg2w2, 128, sg2w3, 128, sg2l, cat1 + 384);
  static_layer(cat1 + 384, 704, 128, sg3w1, 192, sg3w2, 192, sg3w3, 192, sg3l, cat1 + 512);
  dynamic_layer(cat1 + 0, 704, 64, dg1m1, 128, dg1m2, 128, dg1l, cat1 + 64);
  dynamic_layer(cat1 + 64, 704, 128, dg2m1, 192, dg2m2, 192, dg2l, cat1 + 192);

  // f1: BN + lrelu(0.2); result lives in cat2[:, :1024], cat1 copied alongside
  k_gemm<2><<<dim3(N_ / 16, 8), 256, 0, stream>>>(cat1, 704, 704, f1w, 1024, cat2, 1728,
                                                  0.f, nullptr, 0);
  k_bn_stats<<<dim3(1024), 256, 0, stream>>>(cat2, 1728, N_, f1g, f1b, bnsc, bnsh);
  k_bn_apply<<<ew((long long)N_ * 1024), 256, 0, stream>>>(cat2, 1728, 1024,
                                                           (long long)N_ * 1024, bnsc,
                                                           bnsh, 0.2f);
  k_copy_strided<<<ew((long long)N_ * 704), 256, 0, stream>>>(cat1, 704, cat2 + 1024,
                                                              1728, 704,
                                                              (long long)N_ * 704);
  // f2 head
  k_gemm<2><<<dim3(N_ / 16, 4), 256, 0, stream>>>(cat2, 1728, 1728, f2w1, 512, h1, 512,
                                                  0.f, nullptr, 0);
  k_bn_stats<<<dim3(512), 256, 0, stream>>>(h1, 512, N_, f2g1, f2b1, bnsc, bnsh);
  k_bn_apply<<<ew((long long)N_ * 512), 256, 0, stream>>>(h1, 512, 512,
                                                          (long long)N_ * 512, bnsc,
                                                          bnsh, 0.2f);
  k_gemm<2><<<dim3(N_ / 16, 2), 256, 0, stream>>>(h1, 512, 512, f2w2, 256, h2, 256, 0.f,
                                                  nullptr, 0);
  k_bn_stats<<<dim3(256), 256, 0, stream>>>(h2, 256, N_, f2g2, f2b2, bnsc, bnsh);
  k_bn_apply<<<ew((long long)N_ * 256), 256, 0, stream>>>(h2, 256, 256,
                                                          (long long)N_ * 256, bnsc,
                                                          bnsh, 0.2f);
  k_gemm<2><<<dim3(N_ / 16, 1), 256, 0, stream>>>(h2, 256, 256, f2w3, 128, (float*)d_out,
                                                  128, 0.f, nullptr, 0);
}
